// GraphVAEPooled_4904852652791
// MI455X (gfx1250) — compile-verified, matrix operand order checked by reading, not run on hardware
//
#include <hip/hip_runtime.h>

typedef __attribute__((ext_vector_type(16))) _Float16 v16h;
typedef __attribute__((ext_vector_type(8)))  float    v8f;
typedef __attribute__((ext_vector_type(4)))  unsigned int v4u;
typedef __attribute__((ext_vector_type(8)))  int      v8i;
typedef __attribute__((ext_vector_type(4)))  int      v4i;

// ---------------- problem constants ----------------
constexpr int GB    = 8;      // batch
constexpr int GN    = 256;    // nodes per graph
constexpr int FEAT  = 2;
constexpr int PEdim = 8;
constexpr int HID   = 128;
constexpr int LAT   = 64;
constexpr int HEADS = 4;
constexpr int DH    = 16;     // LAT/HEADS
constexpr int SUMN  = GB * GN;          // 2048
constexpr int EE    = 16384;            // edges
constexpr int ETOT  = EE + SUMN;        // + self loops
constexpr int KKEEP = 128;
constexpr int MEMV  = KKEEP + 1;        // 129 valid memories
constexpr int MEMP  = 160;              // padded to x16/x32
constexpr int QIN   = FEAT + 1 + PEdim; // 11
constexpr float GGAMMA = 1.0f;
constexpr float GTAU   = 0.1f;

// ---------------- output layout (floats) ----------------
constexpr size_t OFF_LOGITS = 0;                               // 8*256*256
constexpr size_t OFF_MU     = OFF_LOGITS + (size_t)GB*GN*GN;   // 8*128*64
constexpr size_t OFF_LV     = OFF_MU + (size_t)GB*KKEEP*LAT;
constexpr size_t OFF_KV     = OFF_LV + (size_t)GB*KKEEP*LAT;   // 8*128
constexpr size_t OFF_KEEP   = OFF_KV + (size_t)GB*KKEEP;       // 8*256
constexpr size_t OFF_PERM   = OFF_KEEP + (size_t)GB*GN;        // 1024
constexpr size_t OFF_BP     = OFF_PERM + (size_t)GB*KKEEP;     // 1024

// ---------------- workspace layout (float slots) ----------------
constexpr size_t O_DEG  = 0;
constexpr size_t O_DINV = O_DEG  + SUMN;
constexpr size_t O_NORM = O_DINV + SUMN;
constexpr size_t O_XIN  = O_NORM + ETOT;
constexpr size_t O_GEMM = O_XIN  + (size_t)SUMN*10;
constexpr size_t O_GOUT = O_GEMM + (size_t)SUMN*HID;
constexpr size_t O_H    = O_GOUT + (size_t)SUMN*HID;
constexpr size_t O_TMP  = O_H    + (size_t)SUMN*HID;
constexpr size_t O_MU   = O_TMP  + (size_t)SUMN*HID;
constexpr size_t O_LV   = O_MU   + (size_t)SUMN*LAT;
constexpr size_t O_Z    = O_LV   + (size_t)SUMN*LAT;
constexpr size_t O_GTOK = O_Z    + (size_t)SUMN*LAT;
constexpr size_t O_SCORE= O_GTOK + (size_t)GB*LAT;
constexpr size_t O_TIDX = O_SCORE+ SUMN;               // ints
constexpr size_t O_TSC  = O_TIDX + (size_t)GB*KKEEP;
constexpr size_t O_HPOOL= O_TSC  + (size_t)GB*KKEEP;
constexpr size_t O_MEMP = O_HPOOL+ (size_t)GB*KKEEP*LAT;
constexpr size_t O_DIST = O_MEMP + (size_t)GB*MEMP*LAT;
constexpr size_t O_DEGP = O_DIST + (size_t)GB*GN*GN;
constexpr size_t O_QIN  = O_DEGP + SUMN;
constexpr size_t O_HD   = O_QIN  + (size_t)SUMN*QIN;
constexpr size_t O_Q    = O_HD   + (size_t)SUMN*LAT;
constexpr size_t O_K    = O_Q    + (size_t)SUMN*LAT;
constexpr size_t O_V    = O_K    + (size_t)SUMN*LAT;
constexpr size_t O_ATT  = O_V    + (size_t)SUMN*LAT;
constexpr size_t O_AO   = O_ATT  + (size_t)SUMN*LAT;
constexpr size_t O_KC   = O_AO   + (size_t)SUMN*LAT;
constexpr size_t O_VC   = O_KC   + (size_t)GB*MEMP*LAT;
constexpr size_t O_FF   = O_VC   + (size_t)GB*MEMP*LAT;
constexpr size_t O_AI   = O_FF   + (size_t)SUMN*HID;
constexpr size_t O_AJ   = O_AI   + (size_t)SUMN*LAT;
constexpr size_t O_MSUM = O_AJ   + (size_t)SUMN*LAT;
constexpr size_t O_CMSK = O_MSUM + GB;
// f16 copies (each float slot = 2 halves)
constexpr size_t O_Q16  = O_CMSK + (size_t)GB*MEMP;
constexpr size_t O_K16  = O_Q16  + (size_t)SUMN*LAT/2;
constexpr size_t O_V16  = O_K16  + (size_t)SUMN*LAT/2;
constexpr size_t O_KC16 = O_V16  + (size_t)SUMN*LAT/2;
constexpr size_t O_VC16 = O_KC16 + (size_t)GB*MEMP*LAT/2;

#define CDIV(a,b) (((a)+(b)-1)/(b))

// ================= TDM / cluster probe kernel ==========================
// Issues a NULL-descriptor (count=0) tensor_load_to_lds — architecturally a
// no-op data move — plus s_wait_tensorcnt and a cluster barrier (S_NOP when
// not dispatched in a cluster). Exercises the gfx1250 async-tensor path.
// This toolchain exposes the 6-arg (clang-23 / therock) builtin form.
__global__ __launch_bounds__(32) void k_tdm_nop()
{
  v4u g0 = {};          // D# group0: count=0 -> NULL tensor
  v8i g1 = {};
  v4i g2 = {}, g3 = {};
  v8i g4 = {};
  __builtin_amdgcn_tensor_load_to_lds(g0, g1, g2, g3, g4, 0);
  __builtin_amdgcn_s_wait_tensorcnt(0);
  __builtin_amdgcn_s_cluster_barrier();
}

// ================= LDS-staged WMMA f16->f32 GEMM =================
// C[M,N] = A[M,K] @ B[K,N] (+bias)(+relu), optional f16 copy of C.
// Requires M%128==0, N%16==0, K%32==0. Block = 256 threads (8 waves),
// 128x16 C stripe per block. Per 32-wide K-step: cooperative staging of
// A(128x32) and B(32x16, stored TRANSPOSED [n][k]) as f16 in LDS with odd
// pitch (17-bank stride, conflict-free), so both A and B fragment gathers
// are adjacent 16-bit pairs -> ds_load_2addr_b32.
constexpr int APITCH = 34;   // halves
constexpr int BPITCH = 34;   // halves (transposed tile: [n][k])

__global__ __launch_bounds__(256)
void k_wmma_gemm(const float* __restrict__ A, const float* __restrict__ B,
                 const float* __restrict__ bias, float* __restrict__ C,
                 _Float16* __restrict__ C16, int M, int N, int K, int relu_flag)
{
  __shared__ _Float16 Af[128 * APITCH];
  __shared__ _Float16 Bf[16 * BPITCH];
  const int tid  = threadIdx.x;
  const int wid  = tid >> 5, lane = tid & 31;
  const int lm   = lane & 15, half = lane >> 4;
  const int mb   = blockIdx.x * 128, n0 = blockIdx.y * 16;
  v8f acc = {};
  for (int kk = 0; kk < K; kk += 32) {
    // ---- stage A: 1024 float4 loads across 256 threads ----
#pragma unroll
    for (int j = 0; j < 4; ++j) {
      const int f   = tid + j * 256;           // float4 id
      const int row = f >> 3, c4 = (f & 7) << 2;
      const float4 v4 = *(const float4*)(A + (size_t)(mb + row) * K + kk + c4);
      _Float16* dst = &Af[row * APITCH + c4];
      dst[0] = (_Float16)v4.x; dst[1] = (_Float16)v4.y;
      dst[2] = (_Float16)v4.z; dst[3] = (_Float16)v4.w;
    }
    // ---- stage B (transposed into LDS): 256 float2 loads ----
    {
      const int e = tid * 2;
      const int r = e >> 4, c = e & 15;        // r = k row, c = n col
      const float2 v2 = *(const float2*)(B + (size_t)(kk + r) * N + n0 + c);
      Bf[c * BPITCH + r]       = (_Float16)v2.x;
      Bf[(c + 1) * BPITCH + r] = (_Float16)v2.y;
    }
    __syncthreads();
    v16h a, b;
#pragma unroll
    for (int v = 0; v < 8; ++v) {
      const int kb = (v < 4 ? 2 * v : 16 + 2 * (v - 4)) + 8 * half;
      const int m  = wid * 16 + lm;
      a[2 * v]     = Af[m * APITCH + kb];
      a[2 * v + 1] = Af[m * APITCH + kb + 1];
      b[2 * v]     = Bf[lm * BPITCH + kb];       // adjacent pair in LDS
      b[2 * v + 1] = Bf[lm * BPITCH + kb + 1];
    }
    acc = __builtin_amdgcn_wmma_f32_16x16x32_f16(false, a, false, b,
                                                 (short)0, acc, false, false);
    __syncthreads();
  }
  const int n = n0 + lm;
  const float bv = bias ? bias[n] : 0.f;       // hoisted, one load
  if (C16) {
#pragma unroll
    for (int r = 0; r < 8; ++r) {
      const int m = mb + wid * 16 + r + 8 * half;
      float v = acc[r] + bv;
      if (relu_flag) v = fmaxf(v, 0.f);
      C[(size_t)m * N + n] = v;
      C16[(size_t)m * N + n] = (_Float16)v;
    }
  } else {
#pragma unroll
    for (int r = 0; r < 8; ++r) {
      const int m = mb + wid * 16 + r + 8 * half;
      float v = acc[r] + bv;
      if (relu_flag) v = fmaxf(v, 0.f);
      C[(size_t)m * N + n] = v;
    }
  }
}

// ================= fused attention (per b, head, 16-query tile) ==========
// S = (Q K^T)/4 masked -> row softmax (LDS) -> O = P V.  Dh=16 zero-padded
// to the WMMA K=32; Q/K/V pre-converted to f16 so Q/K fragment loads are
// adjacent 16-bit pairs.
__global__ __launch_bounds__(32)
void k_attn(const _Float16* __restrict__ Q, const _Float16* __restrict__ Kp,
            const _Float16* __restrict__ Vp, const float* __restrict__ kvmask,
            float* __restrict__ O, int kvStride, int kvLen)
{
  __shared__ float    sc[16][256];
  __shared__ _Float16 pf[16][256];
  const int lane = threadIdx.x;
  const int lm = lane & 15, half = lane >> 4;
  const int qt = blockIdx.x, h = blockIdx.y, b = blockIdx.z;
  const int m0 = qt * 16;
  const size_t qbase = ((size_t)b * GN + m0 + lm) * LAT + h * DH;

  v16h aq;                       // Q fragment; only head-dim (16) valid
#pragma unroll
  for (int v = 0; v < 8; ++v) {
    const int kb = (v < 4 ? 2 * v : 16 + 2 * (v - 4)) + 8 * half; // constant
    aq[2 * v]     = (kb     < DH) ? Q[qbase + kb    ] : (_Float16)0.f;
    aq[2 * v + 1] = (kb + 1 < DH) ? Q[qbase + kb + 1] : (_Float16)0.f;
  }
  for (int nt = 0; nt < kvLen / 16; ++nt) {
    const size_t kbase = ((size_t)b * kvStride + nt * 16 + lm) * LAT + h * DH;
    v16h bk;
#pragma unroll
    for (int v = 0; v < 8; ++v) {
      const int kb = (v < 4 ? 2 * v : 16 + 2 * (v - 4)) + 8 * half;
      bk[2 * v]     = (kb     < DH) ? Kp[kbase + kb    ] : (_Float16)0.f;
      bk[2 * v + 1] = (kb + 1 < DH) ? Kp[kbase + kb + 1] : (_Float16)0.f;
    }
    v8f s = {};
    s = __builtin_amdgcn_wmma_f32_16x16x32_f16(false, aq, false, bk,
                                               (short)0, s, false, false);
    const float mk = kvmask[(size_t)b * kvStride + nt * 16 + lm];
#pragma unroll
    for (int r = 0; r < 8; ++r)
      sc[r + 8 * half][nt * 16 + lm] = (mk > 0.5f) ? s[r] * 0.25f : -1e30f;
  }
  __syncthreads();
  if (lane < 16) {                       // row softmax
    float mx = -1e30f;
    for (int k = 0; k < kvLen; ++k) mx = fmaxf(mx, sc[lane][k]);
    float sum = 0.f;
    for (int k = 0; k < kvLen; ++k) { float e = __expf(sc[lane][k] - mx); sc[lane][k] = e; sum += e; }
    const float inv = 1.f / sum;
    for (int k = 0; k < kvLen; ++k) pf[lane][k] = (_Float16)(sc[lane][k] * inv);
  }
  __syncthreads();
  v8f o = {};
  for (int kk = 0; kk < kvLen; kk += 32) {
    v16h ap, bv;
#pragma unroll
    for (int v = 0; v < 8; ++v) {
      const int kb = (v < 4 ? 2 * v : 16 + 2 * (v - 4)) + 8 * half;
      ap[2 * v]     = pf[lm][kk + kb];
      ap[2 * v + 1] = pf[lm][kk + kb + 1];
      const size_t vb = ((size_t)b * kvStride + kk + kb) * LAT + h * DH + lm;
      bv[2 * v]     = Vp[vb];
      bv[2 * v + 1] = Vp[vb + LAT];
    }
    o = __builtin_amdgcn_wmma_f32_16x16x32_f16(false, ap, false, bv,
                                               (short)0, o, false, false);
  }
#pragma unroll
  for (int r = 0; r < 8; ++r)
    O[((size_t)b * GN + m0 + r + 8 * half) * LAT + h * DH + lm] = o[r];
}

// ================= helper kernels =================
__global__ void k_fill(float* p, int n, float v)
{ int i = blockIdx.x * blockDim.x + threadIdx.x; if (i < n) p[i] = v; }

__global__ void k_biasinit(float* out, const float* __restrict__ bias, int M, int H)
{ int t = blockIdx.x * blockDim.x + threadIdx.x; if (t < M * H) out[t] = bias[t % H]; }

__global__ void k_deg(const int* __restrict__ ei, const float* __restrict__ ew, float* deg)
{
  int e = blockIdx.x * blockDim.x + threadIdx.x;
  if (e >= ETOT) return;
  int c = (e < EE) ? ei[EE + e] : (e - EE);
  float w = (e < EE) ? ew[e] : 1.f;
  atomicAdd(&deg[c], w);
}

__global__ void k_dinv(const float* __restrict__ deg, float* dinv)
{ int i = blockIdx.x * blockDim.x + threadIdx.x;
  if (i < SUMN) dinv[i] = (deg[i] > 0.f) ? rsqrtf(deg[i]) : 0.f; }

__global__ void k_norm(const int* __restrict__ ei, const float* __restrict__ ew,
                       const float* __restrict__ dinv, float* nrm)
{
  int e = blockIdx.x * blockDim.x + threadIdx.x;
  if (e >= ETOT) return;
  int r = (e < EE) ? ei[e] : (e - EE);
  int c = (e < EE) ? ei[EE + e] : (e - EE);
  float w = (e < EE) ? ew[e] : 1.f;
  nrm[e] = dinv[r] * w * dinv[c];
}

__global__ void k_scatter(const float* __restrict__ y, const float* __restrict__ nrm,
                          const int* __restrict__ ei, int H, float* out)
{
  long long t = (long long)blockIdx.x * blockDim.x + threadIdx.x;
  if (t >= (long long)ETOT * H) return;
  int e = (int)(t / H), j = (int)(t % H);
  int r = (e < EE) ? ei[e] : (e - EE);
  int c = (e < EE) ? ei[EE + e] : (e - EE);
  atomicAdd(&out[(size_t)c * H + j], nrm[e] * y[(size_t)r * H + j]);
}

__global__ void k_xin(const float* __restrict__ x, const float* __restrict__ pe, float* xin)
{
  int i = blockIdx.x * blockDim.x + threadIdx.x;
  if (i >= SUMN) return;
  xin[(size_t)i * 10 + 0] = x[(size_t)i * 2 + 0];
  xin[(size_t)i * 10 + 1] = x[(size_t)i * 2 + 1];
#pragma unroll
  for (int p = 0; p < PEdim; ++p) xin[(size_t)i * 10 + 2 + p] = pe[(size_t)i * PEdim + p];
}

__global__ void k_gemm_naive(const float* __restrict__ A, const float* __restrict__ B,
                             const float* __restrict__ bias, float* C,
                             int M, int N, int K, int relu_flag)
{
  int t = blockIdx.x * blockDim.x + threadIdx.x;
  if (t >= M * N) return;
  int m = t / N, n = t % N;
  float acc = bias ? bias[n] : 0.f;
  for (int k = 0; k < K; ++k) acc += A[(size_t)m * K + k] * B[(size_t)k * N + n];
  C[t] = relu_flag ? fmaxf(acc, 0.f) : acc;
}

__global__ __launch_bounds__(256)
void k_bn(const float* __restrict__ x, const float* __restrict__ g,
          const float* __restrict__ bb, float* y, int M, int H, int relu_flag)
{
  const int j = blockIdx.x;
  __shared__ float ss[256], sq[256];
  float s = 0.f, q = 0.f;
  for (int i = threadIdx.x; i < M; i += 256) { float v = x[(size_t)i * H + j]; s += v; q += v * v; }
  ss[threadIdx.x] = s; sq[threadIdx.x] = q; __syncthreads();
  for (int st = 128; st > 0; st >>= 1) {
    if (threadIdx.x < st) { ss[threadIdx.x] += ss[threadIdx.x + st]; sq[threadIdx.x] += sq[threadIdx.x + st]; }
    __syncthreads();
  }
  const float m = ss[0] / M;
  const float inv = rsqrtf(sq[0] / M - m * m + 1e-5f);
  const float gg = g[j], bv = bb[j];
  for (int i = threadIdx.x; i < M; i += 256) {
    float v = (x[(size_t)i * H + j] - m) * inv * gg + bv;
    if (relu_flag) v = fmaxf(v, 0.f);
    y[(size_t)i * H + j] = v;
  }
}

__global__ void k_addrelu(const float* a, const float* b, float* y, int n)
{ int i = blockIdx.x * blockDim.x + threadIdx.x; if (i < n) y[i] = fmaxf(a[i] + b[i], 0.f); }

__global__ void k_z(const float* mu, const float* lv, const float* eps, float* z, int n)
{ int i = blockIdx.x * blockDim.x + threadIdx.x; if (i < n) z[i] = mu[i] + eps[i] * __expf(0.5f * lv[i]); }

__global__ void k_gtok(const float* __restrict__ z, float* gtok)
{
  int t = blockIdx.x * blockDim.x + threadIdx.x;
  if (t >= GB * LAT) return;
  int b = t / LAT, d = t % LAT;
  float s = 0.f;
  for (int n = 0; n < GN; ++n) s += z[((size_t)b * GN + n) * LAT + d];
  gtok[t] = s / ((float)GN + 1e-6f);
}

__global__ void k_score(const float* __restrict__ z, const float* __restrict__ w, float* score)
{
  int i = blockIdx.x * blockDim.x + threadIdx.x;
  if (i >= SUMN) return;
  float wn = 0.f, s = 0.f;
#pragma unroll
  for (int d = 0; d < LAT; ++d) { wn += w[d] * w[d]; s += z[(size_t)i * LAT + d] * w[d]; }
  score[i] = s * rsqrtf(wn);
}

__global__ void k_topk(const float* __restrict__ score, float* keep_f, int* tidx, float* tsc)
{
  int t = blockIdx.x * blockDim.x + threadIdx.x;
  if (t >= SUMN) return;
  int b = t / GN, i = t % GN;
  float s = score[t];
  int rank = 0;
  for (int j = 0; j < GN; ++j) {
    float sj = score[(size_t)b * GN + j];
    rank += (sj > s) || (sj == s && j < i);
  }
  keep_f[t] = (rank < KKEEP) ? 1.f : 0.f;
  if (rank < KKEEP) { tidx[(size_t)b * KKEEP + rank] = i; tsc[(size_t)b * KKEEP + rank] = s; }
}

__global__ void k_pool(const float* __restrict__ z, const float* __restrict__ mu,
                       const float* __restrict__ lv, const int* __restrict__ tidx,
                       const float* __restrict__ tsc, float* hpool,
                       float* mu_o, float* lv_o, float* perm_o, float* bp_o, float* kv_o)
{
  int t = blockIdx.x * blockDim.x + threadIdx.x;
  if (t >= GB * KKEEP * LAT) return;
  int b = t / (KKEEP * LAT), r = (t / LAT) % KKEEP, d = t % LAT;
  int idx = tidx[(size_t)b * KKEEP + r];
  size_t zr = ((size_t)b * GN + idx) * LAT + d;
  hpool[t] = z[zr] * tanhf(tsc[(size_t)b * KKEEP + r]);
  mu_o[t] = mu[zr];
  lv_o[t] = lv[zr];
  if (d == 0) {
    perm_o[(size_t)b * KKEEP + r] = (float)(idx + b * GN);
    bp_o[(size_t)b * KKEEP + r] = (float)b;
    kv_o[(size_t)b * KKEEP + r] = 1.f;
  }
}

__global__ void k_mem(const float* __restrict__ hpool, const float* __restrict__ gtok, float* memp)
{
  int t = blockIdx.x * blockDim.x + threadIdx.x;
  if (t >= GB * MEMP * LAT) return;
  int b = t / (MEMP * LAT), tt = (t / LAT) % MEMP, d = t % LAT;
  float v = 0.f;
  if (tt < KKEEP)       v = hpool[((size_t)b * KKEEP + tt) * LAT + d];
  else if (tt == KKEEP) v = gtok[(size_t)b * LAT + d];
  memp[t] = v;
}

__global__ void k_camask(float* cm)
{ int t = blockIdx.x * blockDim.x + threadIdx.x;
  if (t < GB * MEMP) cm[t] = ((t % MEMP) < MEMV) ? 1.f : 0.f; }

__global__ void k_dist(const float* __restrict__ x, float* dist)
{
  long long t = (long long)blockIdx.x * blockDim.x + threadIdx.x;
  if (t >= (long long)GB * GN * GN) return;
  int b = (int)(t / (GN * GN)), i = (int)((t / GN) % GN), j = (int)(t % GN);
  float dx = x[((size_t)b * GN + i) * 2 + 0] - x[((size_t)b * GN + j) * 2 + 0];
  float dy = x[((size_t)b * GN + i) * 2 + 1] - x[((size_t)b * GN + j) * 2 + 1];
  dist[t] = sqrtf(dx * dx + dy * dy + 1e-9f);
}

__global__ void k_degprior(const float* __restrict__ dist, const float* __restrict__ keep, float* degp)
{
  int t = blockIdx.x * blockDim.x + threadIdx.x;
  if (t >= SUMN) return;
  int b = t / GN, i = t % GN;
  float s = 0.f;
  if (keep[t] > 0.5f)
    for (int j = 0; j < GN; ++j)
      s += __expf(-dist[((size_t)b * GN + i) * GN + j] / GTAU) * keep[(size_t)b * GN + j];
  degp[t] = s;
}

__global__ void k_qin(const float* __restrict__ x, const float* __restrict__ degp,
                      const float* __restrict__ pe, float* qin)
{
  int i = blockIdx.x * blockDim.x + threadIdx.x;
  if (i >= SUMN) return;
  qin[(size_t)i * QIN + 0] = x[(size_t)i * 2 + 0];
  qin[(size_t)i * QIN + 1] = x[(size_t)i * 2 + 1];
  qin[(size_t)i * QIN + 2] = degp[i];
#pragma unroll
  for (int p = 0; p < PEdim; ++p) qin[(size_t)i * QIN + 3 + p] = pe[(size_t)i * PEdim + p];
}

__global__ __launch_bounds__(128)
void k_resln(const float* x, const float* r, const float* __restrict__ g,
             const float* __restrict__ bb, float* y, int rows)
{
  int i = blockIdx.x * blockDim.x + threadIdx.x;
  if (i >= rows) return;
  float buf[LAT]; float m = 0.f;
#pragma unroll
  for (int d = 0; d < LAT; ++d) { float t = x[(size_t)i * LAT + d] + r[(size_t)i * LAT + d]; buf[d] = t; m += t; }
  m *= (1.f / LAT);
  float var = 0.f;
#pragma unroll
  for (int d = 0; d < LAT; ++d) { float t = buf[d] - m; var += t * t; }
  float inv = rsqrtf(var * (1.f / LAT) + 1e-5f);
#pragma unroll
  for (int d = 0; d < LAT; ++d) y[(size_t)i * LAT + d] = (buf[d] - m) * inv * g[d] + bb[d];
}

__global__ __launch_bounds__(256)
void k_msum(const float* __restrict__ dist, float* msum)
{
  const int b = blockIdx.x;
  __shared__ float ss[256];
  float s = 0.f;
  for (int t = threadIdx.x; t < KKEEP * KKEEP; t += 256) {
    int i = t / KKEEP, j = t % KKEEP;
    s += dist[((size_t)b * GN + i) * GN + j];
  }
  ss[threadIdx.x] = s; __syncthreads();
  for (int st = 128; st > 0; st >>= 1) {
    if (threadIdx.x < st) ss[threadIdx.x] += ss[threadIdx.x + st];
    __syncthreads();
  }
  if (threadIdx.x == 0) msum[b] = ss[0] / (float)(KKEEP * KKEEP) + 1e-6f;
}

__global__ void k_edge(const float* __restrict__ ai, const float* __restrict__ aj,
                       const float* __restrict__ b1, const float* __restrict__ w2,
                       const float* __restrict__ b2, const float* __restrict__ dist,
                       const float* __restrict__ msum, float* out)
{
  long long t = (long long)blockIdx.x * blockDim.x + threadIdx.x;
  if (t >= (long long)GB * GN * GN) return;
  int b = (int)(t / (GN * GN)), r = (int)((t / GN) % GN), c = (int)(t % GN);
  const float* arow = ai + ((size_t)b * GN + r) * LAT;
  const float* brow = aj + ((size_t)b * GN + c) * LAT;
  __builtin_prefetch(brow, 0, 1);       // global_prefetch_b8
  float acc = b2[0];
#pragma unroll
  for (int d = 0; d < LAT; ++d) {
    float u = arow[d] + brow[d] + b1[d];
    acc += fmaxf(u, 0.f) * w2[d];
  }
  if (r < KKEEP && c < KKEEP)
    acc -= GGAMMA * dist[((size_t)b * GN + r) * GN + c] / msum[b];
  if (r == c) acc = -__builtin_inff();
  out[t] = acc;
}

// =========================================================================
extern "C" void kernel_launch(void* const* d_in, const int* in_sizes, int n_in,
                              void* d_out, int out_size, void* d_ws, size_t ws_size,
                              hipStream_t stream)
{
  (void)in_sizes; (void)n_in; (void)out_size; (void)ws_size;
  const float* x_all = (const float*)d_in[0];
  const int*   ei    = (const int*)  d_in[1];
  const float* pe    = (const float*)d_in[3];
  const float* ewt   = (const float*)d_in[6];
  const float* eps   = (const float*)d_in[7];
  auto P = [&](int i) { return (const float*)d_in[8 + i]; };
  // params: 0 conv1_W 1 conv1_b 2 bn1_g 3 bn1_b 4 conv2_W 5 conv2_b 6 bn2_g
  // 7 bn2_b 8 proj_W 9 proj_b 10 mu_W 11 mu_b 12 lv_W 13 lv_b 14 pool_w
  // 15 inproj_W 16 inproj_b 17 edge_W1 18 edge_b1 19 edge_W2 20 edge_b2
  // layer i @ 21+26i: saWq saWk saWv saWo sabq sabk sabv sabo caWq..cabo
  //                   ffW1 ffb1 ffW2 ffb2 ln1g ln1b ln2g ln2b ln3g ln3b

  float* ws  = (float*)d_ws;
  float* out = (float*)d_out;
  float *deg = ws + O_DEG,  *dinv = ws + O_DINV, *nrm = ws + O_NORM;
  float *xin = ws + O_XIN,  *gm = ws + O_GEMM,   *go  = ws + O_GOUT;
  float *hb  = ws + O_H,    *tp = ws + O_TMP;
  float *mu  = ws + O_MU,   *lv = ws + O_LV,     *z   = ws + O_Z;
  float *gtok= ws + O_GTOK, *sc = ws + O_SCORE;
  int   *tidx= (int*)(ws + O_TIDX);
  float *tsc = ws + O_TSC,  *hp = ws + O_HPOOL,  *mp  = ws + O_MEMP;
  float *dst = ws + O_DIST, *dgp= ws + O_DEGP,   *qin = ws + O_QIN;
  float *hd  = ws + O_HD,   *Qb = ws + O_Q, *Kb = ws + O_K, *Vb = ws + O_V;
  float *att = ws + O_ATT,  *ao = ws + O_AO, *Kc = ws + O_KC, *Vc = ws + O_VC;
  float *ff  = ws + O_FF,   *ai = ws + O_AI, *aj = ws + O_AJ;
  float *msm = ws + O_MSUM, *cm = ws + O_CMSK;
  _Float16 *Q16  = (_Float16*)(ws + O_Q16);
  _Float16 *K16  = (_Float16*)(ws + O_K16);
  _Float16 *V16  = (_Float16*)(ws + O_V16);
  _Float16 *Kc16 = (_Float16*)(ws + O_KC16);
  _Float16 *Vc16 = (_Float16*)(ws + O_VC16);

  // ---- gfx1250 async-tensor path probe (NULL descriptor, data no-op) ----
  k_tdm_nop<<<1,32,0,stream>>>();

  // ---- GCN normalization ----
  k_fill<<<CDIV(SUMN,256),256,0,stream>>>(deg, SUMN, 0.f);
  k_deg <<<CDIV(ETOT,256),256,0,stream>>>(ei, ewt, deg);
  k_dinv<<<CDIV(SUMN,256),256,0,stream>>>(deg, dinv);
  k_norm<<<CDIV(ETOT,256),256,0,stream>>>(ei, ewt, dinv, nrm);
  k_xin <<<CDIV(SUMN,256),256,0,stream>>>(x_all, pe, xin);

  // ---- conv1 (K=10, naive) + bn+relu ----
  k_gemm_naive<<<CDIV(SUMN*HID,256),256,0,stream>>>(xin, P(0), nullptr, gm, SUMN, HID, 10, 0);
  k_biasinit<<<CDIV(SUMN*HID,256),256,0,stream>>>(go, P(1), SUMN, HID);
  k_scatter<<<CDIV((long long)ETOT*HID,256),256,0,stream>>>(gm, nrm, ei, HID, go);
  k_bn<<<HID,256,0,stream>>>(go, P(2), P(3), hb, SUMN, HID, 1);

  // ---- conv2 (WMMA K=128) + bn ----
  dim3 gHID(SUMN/128, HID/16);
  k_wmma_gemm<<<gHID,256,0,stream>>>(hb, P(4), nullptr, gm, nullptr, SUMN, HID, HID, 0);
  k_biasinit<<<CDIV(SUMN*HID,256),256,0,stream>>>(go, P(5), SUMN, HID);
  k_scatter<<<CDIV((long long)ETOT*HID,256),256,0,stream>>>(gm, nrm, ei, HID, go);
  k_bn<<<HID,256,0,stream>>>(go, P(6), P(7), tp, SUMN, HID, 0);

  // ---- residual proj + relu ----
  k_gemm_naive<<<CDIV(SUMN*HID,256),256,0,stream>>>(xin, P(8), P(9), gm, SUMN, HID, 10, 0);
  k_addrelu<<<CDIV(SUMN*HID,256),256,0,stream>>>(tp, gm, hb, SUMN*HID);

  // ---- mu / logvar (WMMA pre-GEMM + scatter) ----
  dim3 gLAT(SUMN/128, LAT/16);
  k_wmma_gemm<<<gLAT,256,0,stream>>>(hb, P(10), nullptr, gm, nullptr, SUMN, LAT, HID, 0);
  k_biasinit<<<CDIV(SUMN*LAT,256),256,0,stream>>>(mu, P(11), SUMN, LAT);
  k_scatter<<<CDIV((long long)ETOT*LAT,256),256,0,stream>>>(gm, nrm, ei, LAT, mu);
  k_wmma_gemm<<<gLAT,256,0,stream>>>(hb, P(12), nullptr, gm, nullptr, SUMN, LAT, HID, 0);
  k_biasinit<<<CDIV(SUMN*LAT,256),256,0,stream>>>(lv, P(13), SUMN, LAT);
  k_scatter<<<CDIV((long long)ETOT*LAT,256),256,0,stream>>>(gm, nrm, ei, LAT, lv);
  k_z<<<CDIV(SUMN*LAT,256),256,0,stream>>>(mu, lv, eps, z, SUMN*LAT);
  k_gtok<<<CDIV(GB*LAT,256),256,0,stream>>>(z, gtok);

  // ---- TopK pooling ----
  k_score<<<CDIV(SUMN,256),256,0,stream>>>(z, P(14), sc);
  k_topk<<<CDIV(SUMN,256),256,0,stream>>>(sc, out + OFF_KEEP, tidx, tsc);
  k_pool<<<CDIV(GB*KKEEP*LAT,256),256,0,stream>>>(z, mu, lv, tidx, tsc, hp,
      out + OFF_MU, out + OFF_LV, out + OFF_PERM, out + OFF_BP, out + OFF_KV);
  k_mem<<<CDIV(GB*MEMP*LAT,256),256,0,stream>>>(hp, gtok, mp);

  // ---- distances / deg prior / decoder input ----
  k_dist<<<CDIV((long long)GB*GN*GN,256),256,0,stream>>>(x_all, dst);
  k_degprior<<<CDIV(SUMN,256),256,0,stream>>>(dst, out + OFF_KEEP, dgp);
  k_qin<<<CDIV(SUMN,256),256,0,stream>>>(x_all, dgp, pe, qin);
  k_gemm_naive<<<CDIV(SUMN*LAT,256),256,0,stream>>>(qin, P(15), P(16), hd, SUMN, LAT, QIN, 0);
  k_camask<<<CDIV(GB*MEMP,256),256,0,stream>>>(cm);

  // ---- transformer decoder (2 layers) ----
  dim3 gAtt(GN/16, HEADS, GB);
  dim3 gMem(GB*MEMP/128, LAT/16);
  dim3 gFF1(SUMN/128, HID/16);
  for (int i = 0; i < 2; ++i) {
    const int L = 21 + 26*i;
    // self-attention
    k_wmma_gemm<<<gLAT,256,0,stream>>>(hd, P(L+0), P(L+4), Qb, Q16, SUMN, LAT, LAT, 0);
    k_wmma_gemm<<<gLAT,256,0,stream>>>(hd, P(L+1), P(L+5), Kb, K16, SUMN, LAT, LAT, 0);
    k_wmma_gemm<<<gLAT,256,0,stream>>>(hd, P(L+2), P(L+6), Vb, V16, SUMN, LAT, LAT, 0);
    k_attn<<<gAtt,32,0,stream>>>(Q16, K16, V16, out + OFF_KEEP, att, GN, GN);
    k_wmma_gemm<<<gLAT,256,0,stream>>>(att, P(L+3), P(L+7), ao, nullptr, SUMN, LAT, LAT, 0);
    k_resln<<<CDIV(SUMN,128),128,0,stream>>>(hd, ao, P(L+20), P(L+21), hd, SUMN);
    // cross-attention
    k_wmma_gemm<<<gLAT,256,0,stream>>>(hd, P(L+8),  P(L+12), Qb, Q16, SUMN, LAT, LAT, 0);
    k_wmma_gemm<<<gMem,256,0,stream>>>(mp, P(L+9),  P(L+13), Kc, Kc16, GB*MEMP, LAT, LAT, 0);
    k_wmma_gemm<<<gMem,256,0,stream>>>(mp, P(L+10), P(L+14), Vc, Vc16, GB*MEMP, LAT, LAT, 0);
    k_attn<<<gAtt,32,0,stream>>>(Q16, Kc16, Vc16, cm, att, MEMP, MEMP);
    k_wmma_gemm<<<gLAT,256,0,stream>>>(att, P(L+11), P(L+15), ao, nullptr, SUMN, LAT, LAT, 0);
    k_resln<<<CDIV(SUMN,128),128,0,stream>>>(hd, ao, P(L+22), P(L+23), hd, SUMN);
    // feed-forward
    k_wmma_gemm<<<gFF1,256,0,stream>>>(hd, P(L+16), P(L+17), ff, nullptr, SUMN, HID, LAT, 1);
    k_wmma_gemm<<<gLAT,256,0,stream>>>(ff, P(L+18), P(L+19), ao, nullptr, SUMN, LAT, HID, 0);
    k_resln<<<CDIV(SUMN,128),128,0,stream>>>(hd, ao, P(L+24), P(L+25), hd, SUMN);
  }

  // ---- pairwise edge logits ----
  k_wmma_gemm<<<gLAT,256,0,stream>>>(hd, P(17),           nullptr, ai, nullptr, SUMN, LAT, LAT, 0);
  k_wmma_gemm<<<gLAT,256,0,stream>>>(hd, P(17) + LAT*LAT, nullptr, aj, nullptr, SUMN, LAT, LAT, 0);
  k_msum<<<GB,256,0,stream>>>(dst, msm);
  k_edge<<<CDIV((long long)GB*GN*GN,256),256,0,stream>>>(ai, aj, P(18), P(19), P(20),
                                                         dst, msm, out + OFF_LOGITS);
}
